// DeltaNet_34007551049775
// MI455X (gfx1250) — matrix operand order repaired and checked by reading, not compiled, not run on hardware
//
#include <hip/hip_runtime.h>
#include <cstdint>
#include <cstddef>

// ---------------------------------------------------------------------------
// Types for CDNA5 WMMA (wave32, 16x16x32 bf16 -> f32)
// ---------------------------------------------------------------------------
typedef __bf16 bf16_t;
typedef __attribute__((ext_vector_type(16))) __bf16 bf16x16;
typedef __attribute__((ext_vector_type(8)))  __bf16 bf16x8;
typedef __attribute__((ext_vector_type(8)))  float  f32x8;
typedef __attribute__((ext_vector_type(4)))  float  f32x4;
typedef __attribute__((ext_vector_type(4)))  unsigned u32x4;
typedef __attribute__((ext_vector_type(2)))  unsigned u32x2;
typedef __attribute__((ext_vector_type(8)))  int      i32x8;
typedef __attribute__((ext_vector_type(4)))  int      i32x4;

union Bf16x16U { bf16x16 v; u32x4 u4[2]; };
union Bf16x8U  { bf16x8  v; u32x4 u4;   bf16_t b[8]; };
union F32x8U   { f32x8   v; f32x4 q[2]; float  f[8]; };
union Bf16x4U  { u32x2   u; bf16_t b[4]; };

// native bf16 converts (lower to hardware cvt ops; RNE)
__device__ __forceinline__ bf16_t f2bf(float f) { return (bf16_t)f; }
__device__ __forceinline__ float bf2f(bf16_t b) { return (float)b; }

__device__ __forceinline__ f32x8 wmma_bf16(bf16x16 a, bf16x16 b, f32x8 c) {
  // D = A(16x32) x B(32x16) + C, fp32 accumulate.
  return __builtin_amdgcn_wmma_f32_16x16x32_bf16(false, a, false, b, (short)0, c,
                                                 false, false);
}

static constexpr int Bc = 2, Lc = 2048, Dc = 2048, Hc = 16, DHc = 128;
static constexpr int Mrows = Bc * Lc;  // 4096

// ---------------------------------------------------------------------------
// Tensor Data Mover (gfx1250): 2D tile load Global -> LDS via D# descriptor.
// Guarded so compilation never regresses if the builtin is absent.
// ---------------------------------------------------------------------------
#if defined(__gfx1250__)
#if __has_builtin(__builtin_amdgcn_tensor_load_to_lds)
#define USE_TDM 1
#pragma message("probe: tensor_load_to_lds AVAILABLE")
#else
#pragma message("probe: tensor_load_to_lds MISSING")
#endif
#endif

#ifdef USE_TDM
__device__ __forceinline__ void tdm_load_2d_bf16(uint64_t gaddr, unsigned lds_off,
                                                 unsigned rows, unsigned rowlen,
                                                 unsigned row_stride) {
  // D# group 0: count=1 | lds_addr | global_addr[56:0] | type=2
  u32x4 g0;
  g0[0] = 1u;
  g0[1] = lds_off;
  g0[2] = (unsigned)(gaddr & 0xFFFFFFFFu);
  g0[3] = (unsigned)((gaddr >> 32) & 0x01FFFFFFu) | (2u << 30);
  // D# group 1: data_size=1 (2B) | tensor_dim0/1 | tile_dim0/1/2 | dim0 stride
  i32x8 g1;
  g1[0] = (int)(1u << 16);                        // data_size = 2 bytes
  g1[1] = (int)((rowlen & 0xFFFFu) << 16);        // tensor_dim0 [15:0]
  g1[2] = (int)(((rowlen >> 16) & 0xFFFFu) | ((rows & 0xFFFFu) << 16));
  g1[3] = (int)((((rows >> 16) & 0xFFFFu)) | ((rowlen & 0xFFFFu) << 16));  // tile_dim0
  g1[4] = (int)(rows & 0xFFFFu);                  // tile_dim1 (tile_dim2 = 0)
  g1[5] = (int)row_stride;                        // tensor_dim0_stride [31:0]
  g1[6] = 0;
  g1[7] = 0;
  i32x4 gz = {0, 0, 0, 0};
#if __clang_major__ >= 23
  i32x8 gz8 = {0, 0, 0, 0, 0, 0, 0, 0};
  __builtin_amdgcn_tensor_load_to_lds(g0, g1, gz, gz, gz8, 0);
#else
  __builtin_amdgcn_tensor_load_to_lds(g0, g1, gz, gz, 0);
#endif
}
#endif

// ---------------------------------------------------------------------------
// Chunked delta rule, chunk T=32, exact math vs sequential recurrence.
// One block (256 thr = 8 waves) per (b,h). S[128,128] fp32 lives in LDS,
// column-major Scm[n*128+d]. Wave w exclusively owns columns [16w,16w+16).
// Per chunk:  Vpred = Kc @ S ; A = Kc Kc^T ; P = tril(Qc Kc^T)      (WMMA)
//             solve Delta_t = beta_t (v_t - Vpred_t - sum_{s<t} A[t,s] Delta_s)
//             O = Qc @ S + P @ Delta ; S += Kc^T @ Delta            (WMMA)
// Chunk tiles are double-buffered and fetched by the Tensor Data Mover:
// wave 0 issues chunk c+1's three tile loads, then s_wait_tensorcnt(3) drains
// only chunk c's transfers, so the DMA streams behind the compute phases.
// ---------------------------------------------------------------------------
// dynamic-LDS byte offsets of the bf16 chunk buffers (dyn base assumed at 0)
static constexpr unsigned kLdsBf16Base = (16384u + 4096u + 1024u + 32u) * 4u;  // 86144
__device__ __forceinline__ unsigned tdm_lds_off(int tensor, int buf) {
  return kLdsBf16Base + (unsigned)tensor * 16384u + (unsigned)buf * 8192u;
}

__global__ __launch_bounds__(256) void delta_kernel(
    const bf16_t* __restrict__ Qh, const bf16_t* __restrict__ Kh,
    const bf16_t* __restrict__ Vh, const float* __restrict__ beta,
    float* __restrict__ Oh) {
  extern __shared__ float smem_f[];
  float* Scm   = smem_f;             // 128*128
  float* sVp   = Scm + 128 * 128;    // 32*128   Vpred[t][d]
  float* sAt   = sVp + 32 * 128;     // 32*32    A[t][s] fp32
  float* sbeta = sAt + 32 * 32;      // 32
  bf16_t* sQb = (bf16_t*)(sbeta + 32);  // [2][32][128]
  bf16_t* sKb = sQb + 2 * 32 * 128;     // [2][32][128]
  bf16_t* sVb = sKb + 2 * 32 * 128;     // [2][32][128]
  bf16_t* sKT = sVb + 2 * 32 * 128;     // [128][32]  K^T
  bf16_t* sDT = sKT + 128 * 32;         // [128][32]  Delta^T (bf16)
  bf16_t* sP  = sDT + 128 * 32;         // [32][32]   masked QK^T (bf16)

  const int tid = threadIdx.x;
  const int lane = tid & 31, w = tid >> 5;
  const int l15 = lane & 15, lh = lane >> 4;
  const int bh = blockIdx.x;
  const size_t base = (size_t)bh * Lc * DHc;
  constexpr int NC = Lc / 32;

#ifdef USE_TDM
  if (w == 0) {  // prologue: chunk 0 streams in while S is zero-filled
    tdm_load_2d_bf16((uint64_t)(uintptr_t)(Qh + base), tdm_lds_off(0, 0), 32u, 128u, 128u);
    tdm_load_2d_bf16((uint64_t)(uintptr_t)(Kh + base), tdm_lds_off(1, 0), 32u, 128u, 128u);
    tdm_load_2d_bf16((uint64_t)(uintptr_t)(Vh + base), tdm_lds_off(2, 0), 32u, 128u, 128u);
  }
#endif

  for (int i = tid; i < 128 * 128; i += 256) Scm[i] = 0.f;
  __syncthreads();

  int cur = 0;
#pragma unroll 1
  for (int c = 0; c < NC; ++c) {
    const int t0 = c * 32;
    const bf16_t* cQ = sQb + cur * 32 * 128;
    const bf16_t* cK = sKb + cur * 32 * 128;
    const bf16_t* cV = sVb + cur * 32 * 128;

    // ---- chunk tiles: TDM double-buffer pipeline (or plain staging) ----
#ifdef USE_TDM
    if (w == 0) {
      if (c + 1 < NC) {
        const size_t nxt = base + (size_t)(t0 + 32) * DHc;
        int nb = cur ^ 1;
        tdm_load_2d_bf16((uint64_t)(uintptr_t)(Qh + nxt), tdm_lds_off(0, nb), 32u, 128u, 128u);
        tdm_load_2d_bf16((uint64_t)(uintptr_t)(Kh + nxt), tdm_lds_off(1, nb), 32u, 128u, 128u);
        tdm_load_2d_bf16((uint64_t)(uintptr_t)(Vh + nxt), tdm_lds_off(2, nb), 32u, 128u, 128u);
        __builtin_amdgcn_s_wait_tensorcnt(3);  // chunk c complete; c+1 in flight
      } else {
        __builtin_amdgcn_s_wait_tensorcnt(0);
      }
    }
#else
    {
      int t = tid >> 3, seg = tid & 7;  // 32 rows x 8 segments of 16 elems
      size_t g = base + (size_t)(t0 + t) * DHc + seg * 16;
      int o = t * 128 + seg * 16;
      *(u32x4*)&((bf16_t*)cQ)[o]     = *(const u32x4*)&Qh[g];
      *(u32x4*)&((bf16_t*)cQ)[o + 8] = *(const u32x4*)&Qh[g + 8];
      *(u32x4*)&((bf16_t*)cK)[o]     = *(const u32x4*)&Kh[g];
      *(u32x4*)&((bf16_t*)cK)[o + 8] = *(const u32x4*)&Kh[g + 8];
      *(u32x4*)&((bf16_t*)cV)[o]     = *(const u32x4*)&Vh[g];
      *(u32x4*)&((bf16_t*)cV)[o + 8] = *(const u32x4*)&Vh[g + 8];
    }
#endif
    if (tid < 32) sbeta[tid] = beta[(size_t)bh * Lc + t0 + tid];
    __syncthreads();

    // ---- build K^T from the chunk's K tile ----
    {
      int t = tid >> 3, seg = tid & 7;
      const bf16_t* p = &cK[t * 128 + seg * 16];
      Bf16x8U a, b2;
      a.u4 = *(const u32x4*)p;
      b2.u4 = *(const u32x4*)(p + 8);
#pragma unroll
      for (int j = 0; j < 8; ++j) {
        sKT[(seg * 16 + j) * 32 + t]     = a.b[j];
        sKT[(seg * 16 + 8 + j) * 32 + t] = b2.b[j];
      }
    }
    __syncthreads();

    // ---- phase 1: Vpred columns [16w,16w+16), plus A (waves 0-3) / P (4-7)
    {
      f32x8 vp[2] = {};
#pragma unroll
      for (int db = 0; db < 4; ++db) {  // K-window db*32..+32 over d
        bf16x16 bS;
        {
          const float* ps = &Scm[(16 * w + l15) * 128 + db * 32 + lh * 16];
#pragma unroll
          for (int jj = 0; jj < 4; ++jj) {
            f32x4 sv = *(const f32x4*)(ps + 4 * jj);
            bS[4 * jj + 0] = f2bf(sv.x);
            bS[4 * jj + 1] = f2bf(sv.y);
            bS[4 * jj + 2] = f2bf(sv.z);
            bS[4 * jj + 3] = f2bf(sv.w);
          }
        }
#pragma unroll
        for (int rb = 0; rb < 2; ++rb) {
          const bf16_t* p = &cK[(16 * rb + l15) * 128 + db * 32 + lh * 16];
          Bf16x16U u; u.u4[0] = *(const u32x4*)p; u.u4[1] = *(const u32x4*)(p + 8);
          vp[rb] = wmma_bf16(u.v, bS, vp[rb]);
        }
      }
#pragma unroll
      for (int rb = 0; rb < 2; ++rb) {
        F32x8U u; u.v = vp[rb];
#pragma unroll
        for (int v = 0; v < 8; ++v)
          sVp[(16 * rb + 8 * lh + v) * 128 + 16 * w + l15] = u.f[v];
      }

      // intra-chunk attention tiles: A = K K^T (w<4), G = Q K^T masked (w>=4)
      {
        const int isG = (w >= 4);
        const int wi = w & 3, ti = wi >> 1, tj = wi & 1;
        const bf16_t* Arows = isG ? cQ : cK;
        f32x8 g = {};
#pragma unroll
        for (int db = 0; db < 4; ++db) {
          const bf16_t* pa = &Arows[(16 * ti + l15) * 128 + db * 32 + lh * 16];
          Bf16x16U ua; ua.u4[0] = *(const u32x4*)pa; ua.u4[1] = *(const u32x4*)(pa + 8);
          const bf16_t* pb = &cK[(16 * tj + l15) * 128 + db * 32 + lh * 16];
          Bf16x16U ub; ub.u4[0] = *(const u32x4*)pb; ub.u4[1] = *(const u32x4*)(pb + 8);
          g = wmma_bf16(ua.v, ub.v, g);
        }
        F32x8U u; u.v = g;
#pragma unroll
        for (int v = 0; v < 8; ++v) {
          int m = 16 * ti + 8 * lh + v, n = 16 * tj + l15;
          if (isG) sP[m * 32 + n] = (n <= m) ? f2bf(u.f[v]) : f2bf(0.f);
          else     sAt[m * 32 + n] = u.f[v];
        }
      }
    }
    __syncthreads();

    // ---- phase 2: sequential solve (threads 0..127 own dimension d) ----
    if (tid < 128) {
      const int d = tid;
      float dh_[32];
#pragma unroll
      for (int t = 0; t < 32; ++t) {
        float acc = sVp[t * 128 + d];
        for (int s = 0; s < t; ++s) acc += sAt[t * 32 + s] * dh_[s];
        float dl = sbeta[t] * (bf2f(cV[t * 128 + d]) - acc);
        dh_[t] = dl;
        sDT[d * 32 + t] = f2bf(dl);
      }
    }
    __syncthreads();

    // ---- phase 3: O (uses old S) then S update; wave-local columns ----
    {
      // Delta^T B-frag for column block w (shared by O and S update)
      Bf16x16U fd;
      {
        const bf16_t* p = &sDT[(16 * w + l15) * 32 + lh * 16];
        fd.u4[0] = *(const u32x4*)p; fd.u4[1] = *(const u32x4*)(p + 8);
      }
      // O = Qc @ S_old + P @ Delta
      f32x8 o[2] = {};
#pragma unroll
      for (int db = 0; db < 4; ++db) {
        bf16x16 bS;
        {
          const float* ps = &Scm[(16 * w + l15) * 128 + db * 32 + lh * 16];
#pragma unroll
          for (int jj = 0; jj < 4; ++jj) {
            f32x4 sv = *(const f32x4*)(ps + 4 * jj);
            bS[4 * jj + 0] = f2bf(sv.x);
            bS[4 * jj + 1] = f2bf(sv.y);
            bS[4 * jj + 2] = f2bf(sv.z);
            bS[4 * jj + 3] = f2bf(sv.w);
          }
        }
#pragma unroll
        for (int rb = 0; rb < 2; ++rb) {
          const bf16_t* p = &cQ[(16 * rb + l15) * 128 + db * 32 + lh * 16];
          Bf16x16U u; u.u4[0] = *(const u32x4*)p; u.u4[1] = *(const u32x4*)(p + 8);
          o[rb] = wmma_bf16(u.v, bS, o[rb]);
        }
      }
#pragma unroll
      for (int rb = 0; rb < 2; ++rb) {
        const bf16_t* p = &sP[(16 * rb + l15) * 32 + lh * 16];
        Bf16x16U u; u.u4[0] = *(const u32x4*)p; u.u4[1] = *(const u32x4*)(p + 8);
        o[rb] = wmma_bf16(u.v, fd.v, o[rb]);
      }
#pragma unroll
      for (int rb = 0; rb < 2; ++rb) {
        F32x8U u; u.v = o[rb];
#pragma unroll
        for (int v = 0; v < 8; ++v) {
          int m = 16 * rb + 8 * lh + v;
          Oh[base + (size_t)(t0 + m) * DHc + 16 * w + l15] = u.f[v];
        }
      }

      // S[:, cols of wave w] += Kc^T @ Delta  (one wmma per 16x16 tile)
#pragma unroll
      for (int db = 0; db < 8; ++db) {
        float* ps = &Scm[(16 * w + l15) * 128 + db * 16 + 8 * lh];
        F32x8U cS; cS.q[0] = *(f32x4*)ps; cS.q[1] = *(f32x4*)(ps + 4);
        const bf16_t* pk = &sKT[(db * 16 + l15) * 32 + lh * 16];
        Bf16x16U u; u.u4[0] = *(const u32x4*)pk; u.u4[1] = *(const u32x4*)(pk + 8);
        cS.v = wmma_bf16(u.v, fd.v, cS.v);
        *(f32x4*)ps = cS.q[0]; *(f32x4*)(ps + 4) = cS.q[1];
      }
    }
    __syncthreads();
    cur ^= 1;
  }
}

// ---------------------------------------------------------------------------
// fp32 -> bf16 elementwise (vec4)
// ---------------------------------------------------------------------------
__global__ __launch_bounds__(256) void cvt_x_kernel(const float* __restrict__ in,
                                                    bf16_t* __restrict__ out, int n4) {
  int i = blockIdx.x * blockDim.x + threadIdx.x;
  if (i >= n4) return;
  f32x4 x = ((const f32x4*)in)[i];
  Bf16x4U r;
  r.b[0] = f2bf(x.x); r.b[1] = f2bf(x.y); r.b[2] = f2bf(x.z); r.b[3] = f2bf(x.w);
  ((u32x2*)out)[i] = r.u;
}

// ---------------------------------------------------------------------------
// Transpose + convert weight: Wt[n*D + k] = bf16(W[k*D + n])  (D = 2048)
// ---------------------------------------------------------------------------
__global__ __launch_bounds__(256) void cvt_wt_kernel(const float* __restrict__ W,
                                                     bf16_t* __restrict__ Wt) {
  __shared__ float tile[32][33];
  int tx = threadIdx.x, ty = threadIdx.y;          // 32 x 8
  int x0 = blockIdx.x * 32, y0 = blockIdx.y * 32;  // x: col n, y: row k
#pragma unroll
  for (int j = 0; j < 32; j += 8)
    tile[ty + j][tx] = W[(size_t)(y0 + ty + j) * Dc + x0 + tx];
  __syncthreads();
#pragma unroll
  for (int j = 0; j < 32; j += 8)
    Wt[(size_t)(x0 + ty + j) * Dc + y0 + tx] = f2bf(tile[tx][ty + j]);
}

// ---------------------------------------------------------------------------
// bf16 WMMA GEMM: C[M,N] = A[M,K] * Bt[N,K]^T   (fp32 out, optional silu)
// Block 256 thr (8 waves), tile 128x128, BK=32, double-buffered LDS:
// next tile's global loads are issued before the WMMA block, stored to the
// alternate buffer after it -> one barrier per K-step, latency behind WMMA.
// ---------------------------------------------------------------------------
__global__ __launch_bounds__(256) void gemm_bf16_kernel(
    const bf16_t* __restrict__ A, const bf16_t* __restrict__ Bt,
    float* __restrict__ C, int M, int N, int K, int mode) {
  constexpr int ASTR = 56;  // padded row stride (bf16), 112B: 16B aligned, bank-spread
  __shared__ bf16_t sA[2][128 * ASTR];
  __shared__ bf16_t sB[2][128 * ASTR];

  const int tid = threadIdx.x;
  const int lane = tid & 31, w = tid >> 5;
  const int wm = w & 3, wn = w >> 2;          // 4x2 wave grid
  const int l15 = lane & 15, lh = lane >> 4;  // lane half
  const int row0 = blockIdx.y * 128, col0 = blockIdx.x * 128;

  // staging indices: two 16B transfers per thread per matrix
  const int r0 = tid >> 2, s0 = tid & 3;
  const int r1 = (tid + 256) >> 2, s1 = (tid + 256) & 3;

  // prologue: stage K-slice 0 into buffer 0
  *(u32x4*)&sA[0][r0 * ASTR + s0 * 8] = *(const u32x4*)&A[(size_t)(row0 + r0) * K + s0 * 8];
  *(u32x4*)&sA[0][r1 * ASTR + s1 * 8] = *(const u32x4*)&A[(size_t)(row0 + r1) * K + s1 * 8];
  *(u32x4*)&sB[0][r0 * ASTR + s0 * 8] = *(const u32x4*)&Bt[(size_t)(col0 + r0) * K + s0 * 8];
  *(u32x4*)&sB[0][r1 * ASTR + s1 * 8] = *(const u32x4*)&Bt[(size_t)(col0 + r1) * K + s1 * 8];
  __syncthreads();

  f32x8 acc[2][4] = {};
  int cur = 0;
#pragma unroll 2
  for (int kb = 0; kb < K; kb += 32) {
    const bool hn = (kb + 32 < K);
    u32x4 na0 = {}, na1 = {}, nb0 = {}, nb1 = {};
    if (hn) {  // issue next tile's global loads (latency hidden by WMMAs)
      na0 = *(const u32x4*)&A[(size_t)(row0 + r0) * K + kb + 32 + s0 * 8];
      na1 = *(const u32x4*)&A[(size_t)(row0 + r1) * K + kb + 32 + s1 * 8];
      nb0 = *(const u32x4*)&Bt[(size_t)(col0 + r0) * K + kb + 32 + s0 * 8];
      nb1 = *(const u32x4*)&Bt[(size_t)(col0 + r1) * K + kb + 32 + s1 * 8];
      if (kb + 96 < K) {  // lowers to global_prefetch_b8
        __builtin_prefetch(&A[(size_t)(row0 + (tid & 127)) * K + kb + 96], 0, 1);
        __builtin_prefetch(&Bt[(size_t)(col0 + (tid & 127)) * K + kb + 96], 0, 1);
      }
    }

    Bf16x16U fa[2], fb[4];
#pragma unroll
    for (int rb = 0; rb < 2; ++rb) {
      // A-frag: row m = l15, 16 contiguous K per lane half
      const bf16_t* p = &sA[cur][(32 * wm + 16 * rb + l15) * ASTR + lh * 16];
      fa[rb].u4[0] = *(const u32x4*)p;
      fa[rb].u4[1] = *(const u32x4*)(p + 8);
    }
#pragma unroll
    for (int cb = 0; cb < 4; ++cb) {
      // B-frag: column n = l15, 16 contiguous K per lane half
      const bf16_t* p = &sB[cur][(64 * wn + 16 * cb + l15) * ASTR + lh * 16];
      fb[cb].u4[0] = *(const u32x4*)p;
      fb[cb].u4[1] = *(const u32x4*)(p + 8);
    }
#pragma unroll
    for (int rb = 0; rb < 2; ++rb)
#pragma unroll
      for (int cb = 0; cb < 4; ++cb)
        acc[rb][cb] = wmma_bf16(fa[rb].v, fb[cb].v, acc[rb][cb]);

    if (hn) {  // store next tile into the alternate buffer
      const int nxt = cur ^ 1;
      *(u32x4*)&sA[nxt][r0 * ASTR + s0 * 8] = na0;
      *(u32x4*)&sA[nxt][r1 * ASTR + s1 * 8] = na1;
      *(u32x4*)&sB[nxt][r0 * ASTR + s0 * 8] = nb0;
      *(u32x4*)&sB[nxt][r1 * ASTR + s1 * 8] = nb1;
    }
    __syncthreads();
    cur ^= 1;
  }

  // epilogue: C layout -> element (m = 8*lh + v, n = l15) per tile
#pragma unroll
  for (int rb = 0; rb < 2; ++rb)
#pragma unroll
    for (int cb = 0; cb < 4; ++cb) {
      F32x8U u; u.v = acc[rb][cb];
#pragma unroll
      for (int v = 0; v < 8; ++v) {
        int rowG = row0 + 32 * wm + 16 * rb + 8 * lh + v;
        int colG = col0 + 64 * wn + 16 * cb + l15;
        float x = u.f[v];
        if (mode == 1) x = x / (1.f + __expf(-x));  // silu
        C[(size_t)rowG * N + colG] = x;
      }
    }
}

// ---------------------------------------------------------------------------
// beta = sigmoid(X @ Wb), layout [B*H, L]
// ---------------------------------------------------------------------------
__global__ __launch_bounds__(256) void beta_kernel(const float* __restrict__ X,
                                                   const float* __restrict__ Wb,
                                                   float* __restrict__ betaO) {
  __shared__ float red[16][17];
  int bl = blockIdx.x;  // 0..4095
  int b = bl >> 11, l = bl & 2047;
  int h = threadIdx.x & 15, slice = threadIdx.x >> 4;
  const float* xr = X + (size_t)bl * Dc + slice * 128;
  float acc = 0.f;
  for (int i = 0; i < 128; ++i) acc += xr[i] * Wb[(size_t)(slice * 128 + i) * Hc + h];
  red[slice][h] = acc;
  __syncthreads();
  if (threadIdx.x < 16) {
    float s = 0.f;
    for (int i = 0; i < 16; ++i) s += red[i][threadIdx.x];
    betaO[(size_t)(b * Hc + threadIdx.x) * Lc + l] = 1.f / (1.f + __expf(-s));
  }
}

// ---------------------------------------------------------------------------
// 128-thread block reduce
// ---------------------------------------------------------------------------
__device__ __forceinline__ float blockReduce128(float v, float* red) {
#pragma unroll
  for (int o = 16; o > 0; o >>= 1) v += __shfl_down(v, o, 32);
  int lane = threadIdx.x & 31, wid = threadIdx.x >> 5;
  __syncthreads();
  if (lane == 0) red[wid] = v;
  __syncthreads();
  return red[0] + red[1] + red[2] + red[3];
}

// l2norm q (+scale), l2norm k, convert v; reorder to [B,H,L,DH] bf16
__global__ __launch_bounds__(128) void prep_kernel(
    const float* __restrict__ Qf, const float* __restrict__ Kf,
    const float* __restrict__ Vf, bf16_t* __restrict__ Qh,
    bf16_t* __restrict__ Kh, bf16_t* __restrict__ Vh) {
  __shared__ float red[4];
  int idx = blockIdx.x;  // bh*2048 + l
  int bh = idx >> 11, l = idx & 2047;
  int b = bh >> 4, h = bh & 15, d = threadIdx.x;
  size_t src = ((size_t)(b * Lc + l)) * Dc + h * DHc + d;
  size_t dst = ((size_t)bh * Lc + l) * DHc + d;
  float q = Qf[src], k = Kf[src], v = Vf[src];
  float nq = blockReduce128(q * q, red);
  float nk = blockReduce128(k * k, red);
  float qn = q / fmaxf(sqrtf(nq), 1e-12f) * 0.08838834764831845f;  // * DH^-0.5
  float kn = k / fmaxf(sqrtf(nk), 1e-12f);
  Qh[dst] = f2bf(qn);
  Kh[dst] = f2bf(kn);
  Vh[dst] = f2bf(v);
}

// RMSNorm over DH, * weight, reorder to [B*L, D] bf16
__global__ __launch_bounds__(128) void rms_kernel(const float* __restrict__ Oh,
                                                  const float* __restrict__ w,
                                                  bf16_t* __restrict__ Obf) {
  __shared__ float red[4];
  int idx = blockIdx.x;
  int bh = idx >> 11, l = idx & 2047;
  int b = bh >> 4, h = bh & 15, d = threadIdx.x;
  float o = Oh[((size_t)bh * Lc + l) * DHc + d];
  float ms = blockReduce128(o * o, red) * (1.f / 128.f);
  float r = o * rsqrtf(ms + 1e-5f) * w[d];
  Obf[((size_t)(b * Lc + l)) * Dc + h * DHc + d] = f2bf(r);
}

// ---------------------------------------------------------------------------
// Host launcher
// ---------------------------------------------------------------------------
extern "C" void kernel_launch(void* const* d_in, const int* in_sizes, int n_in,
                              void* d_out, int out_size, void* d_ws, size_t ws_size,
                              hipStream_t stream) {
  const float* X   = (const float*)d_in[0];
  const float* Wq  = (const float*)d_in[1];
  const float* Wk  = (const float*)d_in[2];
  const float* Wv  = (const float*)d_in[3];
  const float* Wb  = (const float*)d_in[4];
  const float* onw = (const float*)d_in[5];
  const float* Wo  = (const float*)d_in[6];
  float* out = (float*)d_out;

  char* ws = (char*)d_ws;
  size_t off = 0;
  auto carve = [&](size_t bytes) {
    char* p = ws + off;
    off += (bytes + 255) & ~(size_t)255;
    return p;
  };
  bf16_t* Xbf = (bf16_t*)carve((size_t)Mrows * Dc * 2);
  bf16_t* Wqt = (bf16_t*)carve((size_t)Dc * Dc * 2);
  bf16_t* Wkt = (bf16_t*)carve((size_t)Dc * Dc * 2);
  bf16_t* Wvt = (bf16_t*)carve((size_t)Dc * Dc * 2);
  bf16_t* Wot = (bf16_t*)carve((size_t)Dc * Dc * 2);
  float*  Qf  = (float*)carve((size_t)Mrows * Dc * 4);
  float*  Kf  = (float*)carve((size_t)Mrows * Dc * 4);
  float*  Vf  = (float*)carve((size_t)Mrows * Dc * 4);
  float*  bet = (float*)carve((size_t)Bc * Hc * Lc * 4);
  bf16_t* Qh  = (bf16_t*)carve((size_t)Bc * Hc * Lc * DHc * 2);
  bf16_t* Kh  = (bf16_t*)carve((size_t)Bc * Hc * Lc * DHc * 2);
  bf16_t* Vh  = (bf16_t*)carve((size_t)Bc * Hc * Lc * DHc * 2);
  float*  Oh  = (float*)carve((size_t)Bc * Hc * Lc * DHc * 4);
  bf16_t* Obf = (bf16_t*)carve((size_t)Mrows * Dc * 2);

  // 1) conversions
  cvt_x_kernel<<<(Mrows * Dc / 4 + 255) / 256, 256, 0, stream>>>(X, Xbf, Mrows * Dc / 4);
  dim3 tgrid(Dc / 32, Dc / 32), tblk(32, 8);
  cvt_wt_kernel<<<tgrid, tblk, 0, stream>>>(Wq, Wqt);
  cvt_wt_kernel<<<tgrid, tblk, 0, stream>>>(Wk, Wkt);
  cvt_wt_kernel<<<tgrid, tblk, 0, stream>>>(Wv, Wvt);
  cvt_wt_kernel<<<tgrid, tblk, 0, stream>>>(Wo, Wot);

  // 2) projections (WMMA bf16)
  dim3 ggrid(Dc / 128, Mrows / 128);
  gemm_bf16_kernel<<<ggrid, 256, 0, stream>>>(Xbf, Wqt, Qf, Mrows, Dc, Dc, 0);
  gemm_bf16_kernel<<<ggrid, 256, 0, stream>>>(Xbf, Wkt, Kf, Mrows, Dc, Dc, 0);
  gemm_bf16_kernel<<<ggrid, 256, 0, stream>>>(Xbf, Wvt, Vf, Mrows, Dc, Dc, 1);  // silu

  // 3) beta + normalization / reorder
  beta_kernel<<<Bc * Lc, 256, 0, stream>>>(X, Wb, bet);
  prep_kernel<<<Bc * Hc * Lc, 128, 0, stream>>>(Qf, Kf, Vf, Qh, Kh, Vh);

  // 4) chunked delta rule (WMMA + TDM double-buffer); dynamic LDS ~150KB
  size_t dyn = (size_t)(128 * 128 + 32 * 128 + 32 * 32 + 32) * 4 +
               (size_t)(3 * 2 * 32 * 128 + 2 * 128 * 32 + 32 * 32) * 2;
  delta_kernel<<<Bc * Hc, 256, dyn, stream>>>(Qh, Kh, Vh, bet, Oh);

  // 5) RMSNorm + output projection (WMMA)
  rms_kernel<<<Bc * Hc * Lc, 128, 0, stream>>>(Oh, onw, Obf);
  gemm_bf16_kernel<<<ggrid, 256, 0, stream>>>(Obf, Wot, out, Mrows, Dc, Dc, 0);
}
